// GraphEncoder_17446157156478
// MI455X (gfx1250) — compile-verified, matrix operand order checked by reading, not run on hardware
//
#include <hip/hip_runtime.h>
#include <math.h>

// ---------------- problem constants (match reference) ----------------
#define N_NODES 4096
#define N_EDGES 32768
#define NB      32
#define NPG     128
#define EPG     1024
#define EMB     128
#define HID     256
#define NH      4
#define CH      64          // per-head channels (= DH)
#define MOTIF   128
#define E2      (N_EDGES + N_NODES)   // edges + self loops = 36864
#define LN_EPS  1e-5f

typedef __attribute__((ext_vector_type(16))) __bf16 v16bf;
typedef __attribute__((ext_vector_type(8)))  __bf16 v8bf;
typedef __attribute__((ext_vector_type(8)))  float  v8f;
typedef __attribute__((ext_vector_type(4)))  float  v4f;

__device__ __forceinline__ __bf16 f2bf(float f) { return (__bf16)f; }

__device__ __forceinline__ v16bf mk16(v8bf lo, v8bf hi) {
  return __builtin_shufflevector(lo, hi, 0, 1, 2, 3, 4, 5, 6, 7,
                                 8, 9, 10, 11, 12, 13, 14, 15);
}

// Pack 8 consecutive floats (two aligned float4s) into an 8x bf16 vector.
__device__ __forceinline__ v8bf pack8(const float* p) {
  const v4f* q = (const v4f*)p;
  v4f x0 = q[0], x1 = q[1];
  v8bf r;
#pragma unroll
  for (int j = 0; j < 4; ++j) { r[j] = f2bf(x0[j]); r[4 + j] = f2bf(x1[j]); }
  return r;
}

// Build a WMMA 16-bit fragment from a K-contiguous 32-float run in memory/LDS.
// Per-lane elements are runs [half*8, half*8+8) and [16+half*8, 16+half*8+8).
__device__ __forceinline__ v16bf fragf32(const float* base, int half, float scale) {
  const v4f* p = (const v4f*)(base + half * 8);
  const v4f* q = (const v4f*)(base + 16 + half * 8);
  v4f a0 = p[0], a1 = p[1], b0 = q[0], b1 = q[1];
  v16bf r;
#pragma unroll
  for (int e = 0; e < 4; ++e) {
    r[e]      = f2bf(a0[e] * scale);
    r[4 + e]  = f2bf(a1[e] * scale);
    r[8 + e]  = f2bf(b0[e] * scale);
    r[12 + e] = f2bf(b1[e] * scale);
  }
  return r;
}

__device__ __forceinline__ float atomicMaxF(float* addr, float val) {
  unsigned int* a = (unsigned int*)addr;
  unsigned int old = *a;
  while (__uint_as_float(old) < val) {
    unsigned int assumed = old;
    old = atomicCAS(a, assumed, __float_as_uint(val));
    if (old == assumed) break;
  }
  return __uint_as_float(old);
}

// ---------------- generic fill ----------------
__global__ void k_fill(float* p, float v, int n) {
  int i = blockIdx.x * blockDim.x + threadIdx.x;
  if (i < n) p[i] = v;
}

// ---------------- embedding gather ----------------
__global__ void k_gather(const int* __restrict__ t, const float* __restrict__ tab,
                         float* __restrict__ o, int n, int dim) {
  int i = blockIdx.x * blockDim.x + threadIdx.x;
  if (i >= n * dim) return;
  o[i] = tab[(size_t)t[i / dim] * dim + (i % dim)];
}

// ---------------- bf16 WMMA GEMM: C[MxN] = A[MxK] @ W[KxN] (+bias) ----------------
// grid = (N/64, M/64), block = 128 (4 waves); wave w owns rows [w*16, w*16+16),
// all 4 column tiles. M%64==0, K%32==0, N%64==0.
__global__ void k_gemm(const float* __restrict__ A, const float* __restrict__ W,
                       const float* __restrict__ bias, float* __restrict__ C,
                       int M, int K, int Nn) {
  __shared__ __align__(16) __bf16 As[64 * 32];   // [row][k]
  __shared__ __align__(16) __bf16 Bs[64 * 32];   // [n][k]  (transposed tile)
  const int tid  = threadIdx.x;
  const int lane = tid & 31, wave = tid >> 5;
  const int half = lane >> 4, l16 = lane & 15;
  const int m0 = blockIdx.y * 64, n0 = blockIdx.x * 64;
  v8f acc[4] = {{}, {}, {}, {}};
  for (int k0 = 0; k0 < K; k0 += 32) {
    if (k0 + 32 < K) {  // prefetch next K-step tiles into cache
      __builtin_prefetch(&A[(size_t)(m0 + (tid >> 1)) * K + k0 + 32], 0, 1);
      __builtin_prefetch(&W[(size_t)(k0 + 32 + (tid >> 2)) * Nn + n0], 0, 1);
    }
    // A tile: 8 bf16 per store (ds_store_b128), 32B-aligned float4 global loads
    for (int i = tid * 8; i < 2048; i += 1024) {
      int r = i >> 5, c = i & 31;
      *(v8bf*)&As[i] = pack8(&A[(size_t)(m0 + r) * K + k0 + c]);
    }
    // B tile transposed: coalesced float4 over n, scalar stores (transpose)
    for (int i = tid * 4; i < 2048; i += 512) {
      int n = i & 63, kk = i >> 6;
      const v4f x = *(const v4f*)&W[(size_t)(k0 + kk) * Nn + n0 + n];
#pragma unroll
      for (int j = 0; j < 4; ++j) Bs[(n + j) * 32 + kk] = f2bf(x[j]);
    }
    __syncthreads();
    const int arow = wave * 16 + l16;
    v16bf a = mk16(*(const v8bf*)&As[arow * 32 + half * 8],
                   *(const v8bf*)&As[arow * 32 + 16 + half * 8]);
#pragma unroll
    for (int ct = 0; ct < 4; ++ct) {
      const int brow = ct * 16 + l16;
      v16bf b = mk16(*(const v8bf*)&Bs[brow * 32 + half * 8],
                     *(const v8bf*)&Bs[brow * 32 + 16 + half * 8]);
      acc[ct] = __builtin_amdgcn_wmma_f32_16x16x32_bf16(false, a, false, b, (short)0,
                                                        acc[ct], false, false);
    }
    __syncthreads();
  }
#pragma unroll
  for (int ct = 0; ct < 4; ++ct) {
#pragma unroll
    for (int r = 0; r < 8; ++r) {
      int m = m0 + wave * 16 + half * 8 + r;
      int n = n0 + ct * 16 + l16;
      C[(size_t)m * Nn + n] = acc[ct][r] + (bias ? bias[n] : 0.f);
    }
  }
}

// ---------------- GAT self-loop bookkeeping ----------------
__global__ void k_cnt(const int* __restrict__ dsts, float* __restrict__ cnt) {
  int e = blockIdx.x * blockDim.x + threadIdx.x;
  if (e < N_EDGES) atomicAdd(&cnt[dsts[e]], 1.f);
}
__global__ void k_loopacc(const int* __restrict__ dsts, float* __restrict__ ea2) {
  int i = blockIdx.x * blockDim.x + threadIdx.x;
  if (i >= N_EDGES * EMB) return;
  int e = i / EMB, f = i % EMB;
  atomicAdd(&ea2[(size_t)(N_EDGES + dsts[e]) * EMB + f], ea2[(size_t)e * EMB + f]);
}
__global__ void k_loopdiv(const float* __restrict__ cnt, float* __restrict__ ea2) {
  int i = blockIdx.x * blockDim.x + threadIdx.x;
  if (i >= N_NODES * EMB) return;
  int n = i / EMB, f = i % EMB;
  ea2[(size_t)(N_EDGES + n) * EMB + f] *= 1.f / fmaxf(cnt[n], 1.f);
}

// ---------------- GATv2 attention (alpha, softmax, scatter) ----------------
__global__ void k_alpha(const float* __restrict__ xl, const float* __restrict__ xr,
                        const float* __restrict__ ee, const int* __restrict__ srcs,
                        const int* __restrict__ dsts, const float* __restrict__ att,
                        float* __restrict__ alpha, float* __restrict__ amax) {
  int i = blockIdx.x * blockDim.x + threadIdx.x;
  if (i >= E2 * NH) return;
  int e = i / NH, h = i % NH;
  int s, d;
  if (e < N_EDGES) { s = srcs[e]; d = dsts[e]; } else { s = d = e - N_EDGES; }
  const float* pl = xl + (size_t)s * HID + h * CH;
  const float* pr = xr + (size_t)d * HID + h * CH;
  const float* pe = ee + (size_t)e * HID + h * CH;
  const float* pa = att + h * CH;
  float acc = 0.f;
  for (int c = 0; c < CH; ++c) {
    float m = pl[c] + pr[c] + pe[c];
    m = m > 0.f ? m : 0.2f * m;
    acc += m * pa[c];
  }
  alpha[i] = acc;
  atomicMaxF(&amax[d * NH + h], acc);
}

__global__ void k_expden(const int* __restrict__ dsts, const float* __restrict__ amax,
                         float* __restrict__ alpha, float* __restrict__ den) {
  int i = blockIdx.x * blockDim.x + threadIdx.x;
  if (i >= E2 * NH) return;
  int e = i / NH, h = i % NH;
  int d = (e < N_EDGES) ? dsts[e] : (e - N_EDGES);
  float ex = expf(alpha[i] - amax[d * NH + h]);
  alpha[i] = ex;
  atomicAdd(&den[d * NH + h], ex);
}

__global__ void k_scatter(const int* __restrict__ srcs, const int* __restrict__ dsts,
                          const float* __restrict__ alpha, const float* __restrict__ den,
                          const float* __restrict__ xl, float* __restrict__ out) {
  int i = blockIdx.x * blockDim.x + threadIdx.x;
  if (i >= E2 * HID) return;
  int e = i / HID, f = i % HID, h = f / CH;
  int s, d;
  if (e < N_EDGES) { s = srcs[e]; d = dsts[e]; } else { s = d = e - N_EDGES; }
  float a = alpha[(size_t)e * NH + h] / den[d * NH + h];
  atomicAdd(&out[(size_t)d * HID + f], a * xl[(size_t)s * HID + f]);
}

__global__ void k_bias_relu(const float* __restrict__ acc, const float* __restrict__ bias,
                            float* __restrict__ x) {
  int i = blockIdx.x * blockDim.x + threadIdx.x;
  if (i >= N_NODES * HID) return;
  x[i] = fmaxf(acc[i] + bias[i % HID], 0.f);
}

// ---------------- flash attention over all N nodes (one wave per head x 16 queries) ----
__global__ void k_attention(const float* __restrict__ q, const float* __restrict__ kmat,
                            const float* __restrict__ vmat, float* __restrict__ o) {
  __shared__ __align__(16) float  Pt[16][32];   // P tile (C-layout -> A-frag transpose)
  __shared__ __align__(16) __bf16 Vs[64 * 32];  // V chunk transposed [dh][key]
  const int lane = threadIdx.x & 31;
  const int half = lane >> 4, l16 = lane & 15;
  const int q0   = blockIdx.x * 16;
  const int hb   = blockIdx.y * CH;
  // Q fragments, prescaled by 1/sqrt(DH)=0.125
  v16bf aq0 = fragf32(q + (size_t)(q0 + l16) * HID + hb, half, 0.125f);
  v16bf aq1 = fragf32(q + (size_t)(q0 + l16) * HID + hb + 32, half, 0.125f);
  float mrow[8], srow[8];
#pragma unroll
  for (int r = 0; r < 8; ++r) { mrow[r] = -1e30f; srow[r] = 0.f; }
  v8f oc[4] = {{}, {}, {}, {}};

  for (int kc = 0; kc < N_NODES; kc += 32) {
    if (kc + 32 < N_NODES) {  // prefetch next K/V chunk
      __builtin_prefetch(kmat + (size_t)(kc + 32 + lane) * HID + hb, 0, 1);
      __builtin_prefetch(vmat + (size_t)(kc + 32 + lane) * HID + hb, 0, 1);
    }
    // stage V chunk transposed into LDS (coalesced float4 over dh)
    for (int i = lane * 4; i < 2048; i += 128) {
      int n = i & 63, kk = i >> 6;
      const v4f x = *(const v4f*)&vmat[(size_t)(kc + kk) * HID + hb + n];
#pragma unroll
      for (int j = 0; j < 4; ++j) Vs[(n + j) * 32 + kk] = f2bf(x[j]);
    }
    // S = Q @ K^T for 32 keys (two 16-key subtiles)
    v8f sa = {}, sb = {};
    {
      v16bf b0 = fragf32(kmat + (size_t)(kc + l16) * HID + hb, half, 1.f);
      v16bf b1 = fragf32(kmat + (size_t)(kc + l16) * HID + hb + 32, half, 1.f);
      sa = __builtin_amdgcn_wmma_f32_16x16x32_bf16(false, aq0, false, b0, (short)0, sa, false, false);
      sa = __builtin_amdgcn_wmma_f32_16x16x32_bf16(false, aq1, false, b1, (short)0, sa, false, false);
      b0 = fragf32(kmat + (size_t)(kc + 16 + l16) * HID + hb, half, 1.f);
      b1 = fragf32(kmat + (size_t)(kc + 16 + l16) * HID + hb + 32, half, 1.f);
      sb = __builtin_amdgcn_wmma_f32_16x16x32_bf16(false, aq0, false, b0, (short)0, sb, false, false);
      sb = __builtin_amdgcn_wmma_f32_16x16x32_bf16(false, aq1, false, b1, (short)0, sb, false, false);
    }
    // online softmax update; row r of the C tile = query (half*8 + r)
    float scale[8];
#pragma unroll
    for (int r = 0; r < 8; ++r) {
      float tmax = fmaxf(sa[r], sb[r]);
      for (int m = 1; m < 16; m <<= 1) tmax = fmaxf(tmax, __shfl_xor(tmax, m, 32));
      float mn = fmaxf(mrow[r], tmax);
      float pa = expf(sa[r] - mn);
      float pb = expf(sb[r] - mn);
      float ps = pa + pb;
      for (int m = 1; m < 16; m <<= 1) ps += __shfl_xor(ps, m, 32);
      scale[r] = expf(mrow[r] - mn);
      srow[r]  = srow[r] * scale[r] + ps;
      mrow[r]  = mn;
      Pt[half * 8 + r][l16]      = pa;
      Pt[half * 8 + r][16 + l16] = pb;
    }
#pragma unroll
    for (int r = 0; r < 8; ++r) {
      oc[0][r] *= scale[r]; oc[1][r] *= scale[r];
      oc[2][r] *= scale[r]; oc[3][r] *= scale[r];
    }
    __syncthreads();
    // P (16x32, LDS) @ V (32x64, LDS transposed) accumulated into O
    v16bf ap = fragf32(&Pt[l16][0], half, 1.f);
#pragma unroll
    for (int ct = 0; ct < 4; ++ct) {
      const int brow = ct * 16 + l16;
      v16bf bv = mk16(*(const v8bf*)&Vs[brow * 32 + half * 8],
                      *(const v8bf*)&Vs[brow * 32 + 16 + half * 8]);
      oc[ct] = __builtin_amdgcn_wmma_f32_16x16x32_bf16(false, ap, false, bv, (short)0,
                                                       oc[ct], false, false);
    }
    __syncthreads();
  }
#pragma unroll
  for (int r = 0; r < 8; ++r) {
    float inv = 1.f / srow[r];
    int m = q0 + half * 8 + r;
    o[(size_t)m * HID + hb + l16]      = oc[0][r] * inv;
    o[(size_t)m * HID + hb + 16 + l16] = oc[1][r] * inv;
    o[(size_t)m * HID + hb + 32 + l16] = oc[2][r] * inv;
    o[(size_t)m * HID + hb + 48 + l16] = oc[3][r] * inv;
  }
}

// ---------------- pooling / motif helpers ----------------
__global__ void k_pool_sum(const float* __restrict__ a, float* __restrict__ gf) {
  int i = blockIdx.x * blockDim.x + threadIdx.x;
  if (i >= NB * HID) return;
  int b = i / HID, f = i % HID;
  const float* p = a + (size_t)b * NPG * HID + f;
  float s = 0.f;
  for (int n = 0; n < NPG; ++n) s += p[(size_t)n * HID];
  gf[i] = s;
}

__global__ void k_ln_gelu(float* __restrict__ h, const float* __restrict__ g,
                          const float* __restrict__ bb, int rows, int dim) {
  int r = blockIdx.x * blockDim.x + threadIdx.x;
  if (r >= rows) return;
  float* p = h + (size_t)r * dim;
  float mu = 0.f;
  for (int i = 0; i < dim; ++i) mu += p[i];
  mu /= (float)dim;
  float var = 0.f;
  for (int i = 0; i < dim; ++i) { float d = p[i] - mu; var += d * d; }
  var /= (float)dim;
  float inv = rsqrtf(var + LN_EPS);
  for (int i = 0; i < dim; ++i) {
    float y = (p[i] - mu) * inv * g[i] + bb[i];
    p[i] = 0.5f * y * (1.f + erff(y * 0.70710678118f));  // exact-erf GELU
  }
}

__global__ void k_motif_agg(const int* __restrict__ srcs, const int* __restrict__ dsts,
                            const float* __restrict__ h, float* __restrict__ agg) {
  int i = blockIdx.x * blockDim.x + threadIdx.x;
  if (i >= N_EDGES * MOTIF) return;
  int e = i / MOTIF, f = i % MOTIF;
  atomicAdd(&agg[(size_t)srcs[e] * MOTIF + f], h[(size_t)dsts[e] * MOTIF + f]);
}

__global__ void k_concat2(const float* __restrict__ a, const float* __restrict__ b,
                          float* __restrict__ o, int rows, int da, int db) {
  int i = blockIdx.x * blockDim.x + threadIdx.x;
  int dt = da + db;
  if (i >= rows * dt) return;
  int r = i / dt, f = i % dt;
  o[i] = (f < da) ? a[(size_t)r * da + f] : b[(size_t)r * db + (f - da)];
}

__global__ void k_meanmax(const float* __restrict__ h, float* __restrict__ memb) {
  int i = blockIdx.x * blockDim.x + threadIdx.x;
  if (i >= NB * MOTIF) return;
  int b = i / MOTIF, f = i % MOTIF;
  const float* p = h + (size_t)b * NPG * MOTIF + f;
  float s = 0.f, mx = -1e30f;
  for (int n = 0; n < NPG; ++n) { float v = p[(size_t)n * MOTIF]; s += v; mx = fmaxf(mx, v); }
  memb[(size_t)b * 2 * MOTIF + f] = s / (float)NPG;
  memb[(size_t)b * 2 * MOTIF + MOTIF + f] = mx;
}

__global__ void k_small_gemm(const float* __restrict__ A, const float* __restrict__ W,
                             const float* __restrict__ bias, float* __restrict__ C,
                             int M, int K, int Nn) {
  int i = blockIdx.x * blockDim.x + threadIdx.x;
  if (i >= M * Nn) return;
  int m = i / Nn, n = i % Nn;
  float acc = bias ? bias[n] : 0.f;
  for (int k = 0; k < K; ++k) acc += A[(size_t)m * K + k] * W[(size_t)k * Nn + n];
  C[i] = acc;
}

// ---------------- host orchestration ----------------
static inline void gemm(const float* A, const float* W, const float* b, float* C,
                        int M, int K, int Nn, hipStream_t s) {
  k_gemm<<<dim3(Nn / 64, M / 64), dim3(128), 0, s>>>(A, W, b, C, M, K, Nn);
}
#define L1D(K, n, ...)                                                     \
  do { long long _n = (long long)(n);                                      \
       K<<<dim3((unsigned)((_n + 255) / 256)), dim3(256), 0, stream>>>(__VA_ARGS__); } while (0)

extern "C" void kernel_launch(void* const* d_in, const int* in_sizes, int n_in,
                              void* d_out, int out_size, void* d_ws, size_t ws_size,
                              hipStream_t stream) {
  (void)in_sizes; (void)n_in; (void)out_size; (void)ws_size;
  const int* node_types = (const int*)d_in[0];
  const int* edge_types = (const int*)d_in[1];
  const int* srcs = (const int*)d_in[2];
  const int* dsts = srcs + N_EDGES;
  // d_in[3] = batch (graphs are contiguous & equal-sized; unused)
  const float* node_embed = (const float*)d_in[4];
  const float* edge_embed = (const float*)d_in[5];
  auto F = [&](int i) { return (const float*)d_in[i]; };

  float* ws = (float*)d_ws;
  float* X    = ws;                    // N*HID
  float* XL   = ws + 1048576;          // N*HID
  float* XR   = ws + 2097152;          // N*HID
  float* OUTB = ws + 3145728;          // N*HID
  float* Q    = ws + 4194304;          // N*HID
  float* Kb   = ws + 5242880;          // N*HID
  float* Vb   = ws + 6291456;          // N*HID
  float* AO   = ws + 7340032;          // N*HID
  float* EA2  = ws + 8388608;          // E2*EMB
  float* EE   = ws + 13107200;         // E2*HID
  float* ALPH = ws + 22544384;         // E2*NH
  float* AMAX = ws + 22691840;         // N*NH
  float* DEN  = ws + 22708224;         // N*NH
  float* CNT  = ws + 22724608;         // N
  float* Hm   = ws + 22728704;         // N*MOTIF
  float* H2   = ws + 23252992;         // N*MOTIF
  float* AGG  = ws + 23777280;         // N*MOTIF
  float* HCAT = ws + 24301568;         // N*2*MOTIF
  float* GF   = ws + 25350144;         // NB*HID
  float* MEMB = ws + 25358336;         // NB*2*MOTIF
  float* MOTF = ws + 25366528;         // NB*HID
  float* FUSE = ws + 25374720;         // NB*2*HID

  float* out_final = (float*)d_out;                 // [NB, HID]
  float* attn_out  = out_final + NB * HID;          // [N, HID]

  // --- embeddings ---
  L1D(k_gather, N_NODES * EMB, node_types, node_embed, X, N_NODES, EMB);
  L1D(k_gather, N_EDGES * EMB, edge_types, edge_embed, EA2, N_EDGES, EMB);

  // --- self-loop attr = mean of incoming edge attrs (constant across layers) ---
  L1D(k_fill, N_NODES, CNT, 0.f, N_NODES);
  L1D(k_fill, N_NODES * EMB, EA2 + (size_t)N_EDGES * EMB, 0.f, N_NODES * EMB);
  L1D(k_cnt, N_EDGES, dsts, CNT);
  L1D(k_loopacc, N_EDGES * EMB, dsts, EA2);
  L1D(k_loopdiv, N_NODES * EMB, CNT, EA2);

  // --- GATv2 stack ---
  int kin = EMB;
  for (int l = 0; l < 3; ++l) {
    const float* wl   = F(6 + l * 5 + 0);
    const float* wr   = F(6 + l * 5 + 1);
    const float* we   = F(6 + l * 5 + 2);
    const float* att  = F(6 + l * 5 + 3);
    const float* bias = F(6 + l * 5 + 4);
    gemm(X, wl, nullptr, XL, N_NODES, kin, HID, stream);
    gemm(X, wr, nullptr, XR, N_NODES, kin, HID, stream);
    gemm(EA2, we, nullptr, EE, E2, EMB, HID, stream);
    L1D(k_fill, N_NODES * NH, AMAX, -1e30f, N_NODES * NH);
    L1D(k_fill, N_NODES * NH, DEN, 0.f, N_NODES * NH);
    L1D(k_fill, N_NODES * HID, OUTB, 0.f, N_NODES * HID);
    L1D(k_alpha, E2 * NH, XL, XR, EE, srcs, dsts, att, ALPH, AMAX);
    L1D(k_expden, E2 * NH, dsts, AMAX, ALPH, DEN);
    L1D(k_scatter, (long long)E2 * HID, srcs, dsts, ALPH, DEN, XL, OUTB);
    L1D(k_bias_relu, N_NODES * HID, OUTB, bias, X);
    kin = HID;
  }

  // --- full multi-head attention ---
  gemm(X, F(21), F(25), Q, N_NODES, HID, HID, stream);
  gemm(X, F(22), F(26), Kb, N_NODES, HID, HID, stream);
  gemm(X, F(23), F(27), Vb, N_NODES, HID, HID, stream);
  k_attention<<<dim3(N_NODES / 16, NH), dim3(32), 0, stream>>>(Q, Kb, Vb, AO);
  gemm(AO, F(24), F(28), attn_out, N_NODES, HID, HID, stream);

  // --- global add pool ---
  L1D(k_pool_sum, NB * HID, attn_out, GF);

  // --- motif network ---
  gemm(attn_out, F(29), F(30), Hm, N_NODES, HID, MOTIF, stream);
  L1D(k_ln_gelu, N_NODES, Hm, F(31), F(32), N_NODES, MOTIF);
  for (int s = 0; s < 3; ++s) {
    const float* sw  = F(33 + s * 4 + 0);
    const float* sb  = F(33 + s * 4 + 1);
    const float* sg  = F(33 + s * 4 + 2);
    const float* sbb = F(33 + s * 4 + 3);
    L1D(k_fill, N_NODES * MOTIF, AGG, 0.f, N_NODES * MOTIF);
    L1D(k_motif_agg, N_EDGES * MOTIF, srcs, dsts, Hm, AGG);
    L1D(k_concat2, N_NODES * 2 * MOTIF, Hm, AGG, HCAT, N_NODES, MOTIF, MOTIF);
    gemm(HCAT, sw, sb, Hm, N_NODES, 2 * MOTIF, MOTIF, stream);
    L1D(k_ln_gelu, N_NODES, Hm, sg, sbb, N_NODES, MOTIF);
  }
  gemm(Hm, F(45), F(46), H2, N_NODES, MOTIF, MOTIF, stream);
  L1D(k_ln_gelu, N_NODES, H2, F(47), F(48), N_NODES, MOTIF);
  L1D(k_meanmax, NB * MOTIF, H2, MEMB);
  L1D(k_small_gemm, NB * HID, MEMB, F(49), F(50), MOTF, NB, 2 * MOTIF, HID);

  // --- fuse + output projection ---
  L1D(k_concat2, NB * 2 * HID, GF, MOTF, FUSE, NB, HID, HID);
  L1D(k_small_gemm, NB * HID, FUSE, F(51), F(52), out_final, NB, 2 * HID, HID);
}
